// TransitionGNN_74869869904048
// MI455X (gfx1250) — matrix operand order repaired and checked by reading, not compiled, not run on hardware
//
#include <hip/hip_runtime.h>

// ---------------------------------------------------------------------------
// Fused TransitionGNN forward for MI455X (gfx1250, wave32, WMMA).
// states[B,K,D] f32, action[B,K,A] f32, W_edge[E,2D,H], b_edge[E,H],
// W_node[K,F,D], b_node[K,D]  ->  out[B,K,D] f32.
// B=8192 K=10 D=64 H=128 A=16 E=90 F=208.
//
// Single fused kernel: agg[:,i,:] lives in accumulator VGPRs (edges with
// SRC==i are contiguous), so the 377 MB msg tensor is never materialized.
// Edge weights are software-pipelined through registers: next edge's 64 KB
// W tile is in flight (global loads) while the current edge's WMMAs run.
// ---------------------------------------------------------------------------

typedef __bf16 bf16x8  __attribute__((ext_vector_type(8)));
typedef __bf16 bf16x16 __attribute__((ext_vector_type(16)));
typedef float  f32x8   __attribute__((ext_vector_type(8)));

#define BB 8192
#define KN 10
#define DD 64
#define HH 128
#define AA 16
#define EE 90
#define FF 208     // D + H + A
#define FP 224     // F padded to 7 k-steps of 32

#define TILE_B   64
#define NTHREADS 512
#define WE_ELEMS (2 * DD * HH)          // 16384 f32 per edge weight
#define WREGS    (WE_ELEMS / NTHREADS)  // 32 floats/thread in the pipeline

// ---- LDS layout (byte offsets; strides padded +16B to dodge bank conflicts)
#define ST_ROW_STRIDE 144                       // 64 bf16 = 128B + 16 pad
#define ST_NODE_BLK   (TILE_B * ST_ROW_STRIDE)  // 9216 per node
#define SZ_STATES     (KN * ST_NODE_BLK)        // 92160

#define SW_STRIDE_E 272                         // 128 bf16 + pad (edge B op)
#define SW_STRIDE_N 464                         // 224 bf16 + pad (node B op)
#define OFF_W  SZ_STATES
#define SZ_W   (HH * SW_STRIDE_E)               // 34816 (node needs 64*464=29696)

#define SA_ROW_STRIDE 464                       // 224 bf16 + pad (node A op)
#define OFF_A  (OFF_W + SZ_W)
#define SZ_A   (TILE_B * SA_ROW_STRIDE)         // 29696

#define LDS_TOTAL (OFF_A + SZ_A)                // 156672 bytes

__device__ __forceinline__ unsigned short f2bf(float f) {
    unsigned int u = __builtin_bit_cast(unsigned int, f);
    unsigned int r = 0x7FFFu + ((u >> 16) & 1u);   // round-to-nearest-even
    return (unsigned short)((u + r) >> 16);
}

__device__ __forceinline__ float fast_tanh(float x) {
    // tanh(x) = (e^{2x}-1)/(e^{2x}+1), e^{2x} = exp2(x * 2*log2(e))
    float e = __builtin_amdgcn_exp2f(x * 2.8853900817779268f);
    return (e - 1.0f) * __builtin_amdgcn_rcpf(e + 1.0f);
}

// A-operand fragment: lane holds two 8-element K runs at +0 and +32 bytes.
__device__ __forceinline__ bf16x16 load_frag_gap32(const unsigned char* p) {
    bf16x8 lo = *(const bf16x8*)(p);
    bf16x8 hi = *(const bf16x8*)(p + 32);
    return __builtin_shufflevector(lo, hi, 0,1,2,3,4,5,6,7,8,9,10,11,12,13,14,15);
}

// Unused here (B operands use load_frag_gap32): lane would hold 16 contiguous K values (32 bytes).
__device__ __forceinline__ bf16x16 load_frag_contig(const unsigned char* p) {
    bf16x8 lo = *(const bf16x8*)(p);
    bf16x8 hi = *(const bf16x8*)(p + 16);
    return __builtin_shufflevector(lo, hi, 0,1,2,3,4,5,6,7,8,9,10,11,12,13,14,15);
}

__global__ __launch_bounds__(NTHREADS, 1)
void gnn_fused_kernel(const float* __restrict__ states,
                      const float* __restrict__ action,
                      const float* __restrict__ W_edge,
                      const float* __restrict__ b_edge,
                      const float* __restrict__ W_node,
                      const float* __restrict__ b_node,
                      float* __restrict__ out) {
    __shared__ __align__(16) unsigned char smem[LDS_TOTAL];
    unsigned char* sStates = smem;
    unsigned char* sW      = smem + OFF_W;
    unsigned char* sA      = smem + OFF_A;

    const int tid   = threadIdx.x;
    const int lane  = tid & 31;
    const int wave  = tid >> 5;
    const int half  = lane >> 4;    // which 16-lane half
    const int l15   = lane & 15;
    const int ntile = wave & 7;     // N-tile of H (8 x 16 = 128)
    const int mpair = wave >> 3;    // this wave owns M-tiles {2*mpair, 2*mpair+1}
    const int b0    = blockIdx.x * TILE_B;

    // ---- stage states tile [64 rows][10 nodes][64] -> bf16 LDS (node-major)
    for (int idx = tid; idx < TILE_B * KN * DD; idx += NTHREADS) {
        int row = idx / (KN * DD);
        int rem = idx - row * (KN * DD);
        int node = rem >> 6;
        int col  = rem & 63;
        float v = states[(size_t)(b0 + row) * (KN * DD) + rem];
        *(unsigned short*)(sStates + node * ST_NODE_BLK + row * ST_ROW_STRIDE + col * 2) = f2bf(v);
    }

    // ---- prime the W_edge register pipeline with edge 0
    float wreg[WREGS];
    #pragma unroll
    for (int t = 0; t < WREGS; ++t)
        wreg[t] = W_edge[(size_t)tid + t * NTHREADS];

    __syncthreads();

    for (int i = 0; i < KN; ++i) {
        // agg[:, i, :] lives entirely in registers: 2 M-tiles x (16x16) per wave
        f32x8 agg[2];
        #pragma unroll
        for (int m = 0; m < 2; ++m) {
            #pragma unroll
            for (int r = 0; r < 8; ++r) agg[m][r] = 0.0f;
        }

        // ----- edge phase: the 9 edges with SRC == i are e = i*9 + jj
        for (int jj = 0; jj < KN - 1; ++jj) {
            const int e = i * (KN - 1) + jj;
            const int j = jj + (jj >= i ? 1 : 0);

            __syncthreads();  // previous consumers of sW are done

            // convert+store the in-register W tile (transposed to B layout)
            #pragma unroll
            for (int t = 0; t < WREGS; ++t) {
                int idx = tid + t * NTHREADS;
                int k = idx >> 7;     // 0..127 (2D dim)
                int n = idx & 127;    // 0..127 (H dim)
                *(unsigned short*)(sW + n * SW_STRIDE_E + k * 2) = f2bf(wreg[t]);
            }
            // issue global loads for the NEXT edge; they overlap the WMMAs
            if (e + 1 < EE) {
                const float* WeN = W_edge + (size_t)(e + 1) * WE_ELEMS;
                #pragma unroll
                for (int t = 0; t < WREGS; ++t)
                    wreg[t] = WeN[tid + t * NTHREADS];
            }
            __syncthreads();

            // B fragments shared across both M-tiles
            bf16x16 bfr[4];
            #pragma unroll
            for (int ks = 0; ks < 4; ++ks)
                bfr[ks] = load_frag_gap32(sW + (ntile * 16 + l15) * SW_STRIDE_E
                                              + ks * 64 + half * 16);
            const float bias = b_edge[e * HH + ntile * 16 + l15];

            #pragma unroll
            for (int m = 0; m < 2; ++m) {
                const int mt = mpair * 2 + m;
                f32x8 c = {0.f,0.f,0.f,0.f,0.f,0.f,0.f,0.f};
                #pragma unroll
                for (int ks = 0; ks < 4; ++ks) {
                    // pair_feat = [states[:,i,:], states[:,j,:]] : pick node block per k-step
                    const int nd = (ks < 2) ? i : j;
                    const unsigned char* pa = sStates + nd * ST_NODE_BLK
                                            + (mt * 16 + l15) * ST_ROW_STRIDE
                                            + (ks & 1) * 64 + half * 16;
                    bf16x16 a = load_frag_gap32(pa);
                    c = __builtin_amdgcn_wmma_f32_16x16x32_bf16(
                            false, a, false, bfr[ks], (short)0, c, false, false);
                }
                #pragma unroll
                for (int r = 0; r < 8; ++r) agg[m][r] += fast_tanh(c[r] + bias);
            }
        }

        // ----- node phase for node i
        __syncthreads();  // all edge WMMAs done; safe to rebuild sW / write sA

        // node_in cols 0..63 = states[:,i,:] (bf16 copy from sStates)
        for (int idx = tid; idx < TILE_B * DD; idx += NTHREADS) {
            int row = idx >> 6, col = idx & 63;
            *(unsigned short*)(sA + row * SA_ROW_STRIDE + col * 2) =
                *(const unsigned short*)(sStates + i * ST_NODE_BLK
                                                 + row * ST_ROW_STRIDE + col * 2);
        }
        // cols 64..79 = action[:,i,:]
        for (int idx = tid; idx < TILE_B * AA; idx += NTHREADS) {
            int row = idx >> 4, col = idx & 15;
            float v = action[(size_t)(b0 + row) * (KN * AA) + i * AA + col];
            *(unsigned short*)(sA + row * SA_ROW_STRIDE + (DD + col) * 2) = f2bf(v);
        }
        // cols 208..223 = zero pad
        for (int idx = tid; idx < TILE_B * (FP - FF); idx += NTHREADS) {
            int row = idx >> 4, col = FF + (idx & 15);
            *(unsigned short*)(sA + row * SA_ROW_STRIDE + col * 2) = 0;
        }
        // cols 80..207 = agg (each wave spills its register tile)
        #pragma unroll
        for (int m = 0; m < 2; ++m) {
            const int mt = mpair * 2 + m;
            #pragma unroll
            for (int r = 0; r < 8; ++r) {
                int row = mt * 16 + r + half * 8;            // C/D layout
                int col = DD + AA + ntile * 16 + l15;
                *(unsigned short*)(sA + row * SA_ROW_STRIDE + col * 2) = f2bf(agg[m][r]);
            }
        }
        // W_node[i]: [208][64] f32 -> transposed bf16 [n][k], k padded to 224
        const float* Wn = W_node + (size_t)i * FF * DD;
        for (int idx = tid; idx < FF * DD; idx += NTHREADS) {
            int k = idx >> 6, n = idx & 63;
            *(unsigned short*)(sW + n * SW_STRIDE_N + k * 2) = f2bf(Wn[idx]);
        }
        for (int idx = tid; idx < DD * (FP - FF); idx += NTHREADS) {
            int n = idx >> 4, k = FF + (idx & 15);
            *(unsigned short*)(sW + n * SW_STRIDE_N + k * 2) = 0;
        }
        __syncthreads();

        // 8 waves (ntile<4) compute the [64 x 64] node output; branch is
        // wave-uniform so EXEC stays all-ones inside (WMMA requirement).
        if (ntile < 4) {
            const float bias = b_node[i * DD + ntile * 16 + l15];
            #pragma unroll
            for (int m = 0; m < 2; ++m) {
                const int mt = mpair * 2 + m;
                f32x8 c = {0.f,0.f,0.f,0.f,0.f,0.f,0.f,0.f};
                #pragma unroll
                for (int ks = 0; ks < 7; ++ks) {
                    bf16x16 a = load_frag_gap32(sA + (mt * 16 + l15) * SA_ROW_STRIDE
                                                   + ks * 64 + half * 16);
                    bf16x16 b = load_frag_gap32(sW + (ntile * 16 + l15) * SW_STRIDE_N
                                                    + ks * 64 + half * 16);
                    c = __builtin_amdgcn_wmma_f32_16x16x32_bf16(
                            false, a, false, b, (short)0, c, false, false);
                }
                #pragma unroll
                for (int r = 0; r < 8; ++r) {
                    int row = mt * 16 + r + half * 8;
                    int col = ntile * 16 + l15;
                    out[(size_t)(b0 + row) * (KN * DD) + i * DD + col] =
                        fast_tanh(c[r] + bias);
                }
            }
        }
    }
}

extern "C" void kernel_launch(void* const* d_in, const int* in_sizes, int n_in,
                              void* d_out, int out_size, void* d_ws, size_t ws_size,
                              hipStream_t stream) {
    const float* states = (const float*)d_in[0];
    const float* action = (const float*)d_in[1];
    const float* W_edge = (const float*)d_in[2];
    const float* b_edge = (const float*)d_in[3];
    const float* W_node = (const float*)d_in[4];
    const float* b_node = (const float*)d_in[5];
    float* out = (float*)d_out;

    dim3 grid(BB / TILE_B);   // 128 workgroups
    dim3 block(NTHREADS);     // 16 waves (wave32)
    hipLaunchKernelGGL(gnn_fused_kernel, grid, block, 0, stream,
                       states, action, W_edge, b_edge, W_node, b_node, out);
}
